// AttentionSW_35003983462679
// MI455X (gfx1250) — matrix operand order repaired and checked by reading, not compiled, not run on hardware
//
#include <hip/hip_runtime.h>

// ------------------------------------------------------------------
// CDNA5 (gfx1250) sliding-window GQA attention, bf16 WMMA everywhere.
// ------------------------------------------------------------------

typedef __attribute__((ext_vector_type(16))) __bf16        v16bf;
typedef __attribute__((ext_vector_type(8)))  float         v8f;
typedef __attribute__((ext_vector_type(4)))  unsigned int  v4u;

union Frag16 {
  v4u            q[2];   // 2 x 16B
  unsigned short s[16];  // 16 bf16 (as raw ushort)
  v16bf          v;
};

__device__ __forceinline__ unsigned short f2bf(float f) {
  unsigned int u = __float_as_uint(f);
  u += 0x7FFFu + ((u >> 16) & 1u);   // round-to-nearest-even
  return (unsigned short)(u >> 16);
}

// ------------------------- elementwise ----------------------------

__global__ void cast_bf16_kernel(const float* __restrict__ src,
                                 unsigned short* __restrict__ dst, int n) {
  int i = blockIdx.x * blockDim.x + threadIdx.x;
  if (i < n) dst[i] = f2bf(src[i]);
}

// src [B, L, nh*128] f32  ->  dst [B, nh, L, 128] bf16, optional RoPE
__global__ void rope_perm_kernel(const float* __restrict__ src,
                                 const float* __restrict__ cosb,
                                 const float* __restrict__ sinb,
                                 unsigned short* __restrict__ dst,
                                 int L, int nh, int n, int do_rope) {
  int idx = blockIdx.x * blockDim.x + threadIdx.x;
  if (idx >= n) return;
  int d  = idx & 127;
  int t  = idx >> 7;
  int l  = t % L;
  int t2 = t / L;
  int h  = t2 % nh;
  int b  = t2 / nh;
  size_t srow = ((size_t)(b * L + l)) * (size_t)(nh * 128) + (size_t)h * 128;
  float v = src[srow + d];
  float res = v;
  if (do_rope) {
    float o = (d < 64) ? -src[srow + d + 64] : src[srow + d - 64];
    res = v * cosb[l * 128 + d] + o * sinb[l * 128 + d];
  }
  dst[idx] = f2bf(res);
}

// --------------------------- GEMM ---------------------------------
// C[M,N] (f32) = A[M,K] (bf16) * W[N,K]^T (bf16).
// One wave per 16x128 output tile (8 accumulators); K stepped by 32
// with v_wmma_f32_16x16x32_bf16. A fragment reused across 8 N-tiles;
// W re-reads are served out of the 192 MB L2.
__global__ __launch_bounds__(32)
void gemm_bf16_nt(const unsigned short* __restrict__ A,
                  const unsigned short* __restrict__ W,
                  float* __restrict__ C, int M, int N, int K) {
  const int lane = threadIdx.x;
  const int half = lane >> 4;   // 0: lanes 0-15, 1: lanes 16-31
  const int l15  = lane & 15;
  const int m0 = blockIdx.y * 16;
  const int n0 = blockIdx.x * 128;

  v8f acc[8] = {};
  const unsigned short* arow = A + (size_t)(m0 + l15) * K + half * 8;

  for (int k0 = 0; k0 < K; k0 += 32) {
    Frag16 a;
    // A 16x32 layout: lane=M, elems 0-7 -> K=half*8+e, elems 8-15 -> +16
    a.q[0] = *(const v4u*)(arow + k0);
    a.q[1] = *(const v4u*)(arow + k0 + 16);
#pragma unroll
    for (int t = 0; t < 8; ++t) {
      // B 32x16 layout: lane=N, elems 0-7 -> K=half*8+e, elems 8-15 -> +16  (row of W = col of B)
      const unsigned short* wrow =
          W + (size_t)(n0 + t * 16 + l15) * K + k0 + half * 8;
      Frag16 b;
      b.q[0] = *(const v4u*)(wrow);
      b.q[1] = *(const v4u*)(wrow + 16);
      acc[t] = __builtin_amdgcn_wmma_f32_16x16x32_bf16(
          false, a.v, false, b.v, (short)0, acc[t], false, false);
    }
  }
  // C/D layout: VGPR r -> row m0 + r + half*8, lane&15 -> col
#pragma unroll
  for (int t = 0; t < 8; ++t)
#pragma unroll
    for (int r = 0; r < 8; ++r)
      C[(size_t)(m0 + r + half * 8) * N + n0 + t * 16 + l15] = acc[t][r];
}

// ----------------------- attention --------------------------------
// One wave per (b, h, 16-query block). Flash-style over key blocks of 32.
// WINDOW=192 -> key blocks [max(0, qb-12) .. qb], <= 13 blocks -> <=7 pairs.
#define HD 128
__global__ __launch_bounds__(32)
void attn_sw_kernel(const unsigned short* __restrict__ Qb,
                    const unsigned short* __restrict__ Kb,
                    const unsigned short* __restrict__ Vb,
                    unsigned short* __restrict__ Ob, int L) {
  __shared__ alignas(16) unsigned short Plds[16 * 32];   // P tile (16q x 32k)
  __shared__ alignas(16) unsigned short Vt[32 * HD];     // V tile (32k x 128d)

  const int lane = threadIdx.x;
  const int half = lane >> 4;
  const int l15  = lane & 15;
  const int qb = blockIdx.x, h = blockIdx.y, b = blockIdx.z;
  const int kvh = h >> 2;                     // n_rep = 4
  const int q0  = qb * 16;
  const size_t qbase  = ((size_t)(b * 16 + h)   * L) * HD;
  const size_t kvbase = ((size_t)(b * 4  + kvh) * L) * HD;

  // Q fragments: 16x128 bf16 = 4 x (16x32) A-fragments
  Frag16 qf[4];
  const unsigned short* qrow = Qb + qbase + (size_t)(q0 + l15) * HD + half * 8;
#pragma unroll
  for (int dd = 0; dd < 4; ++dd) {
    qf[dd].q[0] = *(const v4u*)(qrow + dd * 32);
    qf[dd].q[1] = *(const v4u*)(qrow + dd * 32 + 16);
  }

  float mrow[8], lrow[8];
  v8f   O[8] = {};
#pragma unroll
  for (int r = 0; r < 8; ++r) { mrow[r] = -1e30f; lrow[r] = 0.0f; }

  const int kstart = max(0, qb - 12);
  const int nblk   = qb - kstart + 1;
  const int npair  = (nblk + 1) >> 1;
  const float scale = 0.08838834764831845f;   // 1/sqrt(128)

  // per-lane LDS destination for async V staging (this lane's V row)
  const unsigned int vlds = (unsigned int)(unsigned long long)(Vt + lane * HD);

  for (int p = 0; p < npair; ++p) {
    const int k0 = kstart * 16 + p * 32;

    // ---- stage V tile [32 keys x 128 d] into LDS with CDNA5 async copy.
    // Out-of-range rows are clamped; their P entries are exactly 0 so the
    // (finite) duplicated data never contributes.
    {
      int j = min(k0 + lane, L - 1);
      unsigned long long ga =
          (unsigned long long)(Vb + kvbase + (size_t)j * HD);
#pragma unroll
      for (int t = 0; t < 16; ++t) {
        asm volatile("global_load_async_to_lds_b128 %0, %1, off"
                     :
                     : "v"(vlds + (unsigned)(t * 16)),
                       "v"(ga + (unsigned long long)(t * 16))
                     : "memory");
      }
    }

    // S = Q * K^T  (two 16x16 tiles, 4 WMMAs each over d)
    v8f S[2] = {};
#pragma unroll
    for (int kk = 0; kk < 2; ++kk) {
      int j = min(k0 + kk * 16 + l15, L - 1);   // clamped; mask fixes tail
      const unsigned short* krow = Kb + kvbase + (size_t)j * HD + half * 8;
#pragma unroll
      for (int dd = 0; dd < 4; ++dd) {
        Frag16 kf;
        kf.q[0] = *(const v4u*)(krow + dd * 32);
        kf.q[1] = *(const v4u*)(krow + dd * 32 + 16);
        S[kk] = __builtin_amdgcn_wmma_f32_16x16x32_bf16(
            false, qf[dd].v, false, kf.v, (short)0, S[kk], false, false);
      }
    }

    // mask + online softmax (row = r + half*8 across each 16-lane group)
#pragma unroll
    for (int r = 0; r < 8; ++r) {
      int i  = q0 + r + half * 8;
      int j0 = k0 + l15;                        // true (unclamped) indices
      int j1 = j0 + 16;
      float s0 = S[0][r] * scale;
      float s1 = S[1][r] * scale;
      if (j0 > i || (i - j0) > 192) s0 = -1e30f;
      if (j1 > i || (i - j1) > 192) s1 = -1e30f;
      float vmax = fmaxf(s0, s1);
#pragma unroll
      for (int mk = 8; mk >= 1; mk >>= 1)
        vmax = fmaxf(vmax, __shfl_xor(vmax, mk, 32));
      float mnew  = fmaxf(mrow[r], vmax);
      float alpha = __expf(mrow[r] - mnew);
      float p0 = __expf(s0 - mnew);
      float p1 = __expf(s1 - mnew);
      float ps = p0 + p1;
#pragma unroll
      for (int mk = 8; mk >= 1; mk >>= 1)
        ps += __shfl_xor(ps, mk, 32);
      lrow[r] = lrow[r] * alpha + ps;
      mrow[r] = mnew;
#pragma unroll
      for (int nb = 0; nb < 8; ++nb) O[nb][r] *= alpha;
      int prow = r + half * 8;
      Plds[prow * 32 + l15]      = f2bf(p0);
      Plds[prow * 32 + 16 + l15] = f2bf(p1);
    }
    __syncthreads();

    // P tile -> A fragment (16q x 32k), via LDS b128 reads
    Frag16 pf;
    {
      const unsigned short* prow = Plds + l15 * 32 + half * 8;
      pf.q[0] = *(const v4u*)(prow);
      pf.q[1] = *(const v4u*)(prow + 16);
    }

    // async V staging must have landed before we read Vt
    asm volatile("s_wait_asynccnt 0" ::: "memory");

    // O += P * V  (8 x 16-wide d-blocks)
#pragma unroll
    for (int nb = 0; nb < 8; ++nb) {
      Frag16 vf;   // B 32x16: lane = d column, elems 0-7 -> key = half*8 + e, elems 8-15 -> +16
#pragma unroll
      for (int e = 0; e < 16; ++e)
        vf.s[e] = Vt[((e & 7) + half * 8 + (e >> 3) * 16) * HD + nb * 16 + l15];
      O[nb] = __builtin_amdgcn_wmma_f32_16x16x32_bf16(
          false, pf.v, false, vf.v, (short)0, O[nb], false, false);
    }
    __syncthreads();
  }

  // normalize + store bf16 into [B, L, H*128]
#pragma unroll
  for (int r = 0; r < 8; ++r) {
    float inv = 1.0f / lrow[r];
    int row = q0 + r + half * 8;
    size_t obase = ((size_t)(b * L + row)) * 2048 + (size_t)h * HD;
#pragma unroll
    for (int nb = 0; nb < 8; ++nb)
      Ob[obase + nb * 16 + l15] = f2bf(O[nb][r] * inv);
  }
}

// --------------------------- host ---------------------------------

extern "C" void kernel_launch(void* const* d_in, const int* in_sizes, int n_in,
                              void* d_out, int out_size, void* d_ws, size_t ws_size,
                              hipStream_t stream) {
  (void)in_sizes; (void)n_in; (void)out_size; (void)ws_size;
  const float* x    = (const float*)d_in[0];
  const float* cosb = (const float*)d_in[1];
  const float* sinb = (const float*)d_in[2];
  const float* wq   = (const float*)d_in[3];
  const float* wk   = (const float*)d_in[4];
  const float* wv   = (const float*)d_in[5];
  const float* wo   = (const float*)d_in[6];
  float* out = (float*)d_out;

  const int B = 2, L = 2048, DIM = 2048, H = 16, KVH = 4;
  const int M = B * L;                       // 4096

  char* ws = (char*)d_ws;
  size_t off = 0;
  auto carve = [&](size_t bytes) -> char* {
    char* p = ws + off;
    off += (bytes + 255) & ~(size_t)255;
    return p;
  };
  unsigned short* xb  = (unsigned short*)carve((size_t)M * DIM * 2);
  unsigned short* wqb = (unsigned short*)carve((size_t)DIM * DIM * 2);
  unsigned short* wkb = (unsigned short*)carve((size_t)(KVH * 128) * DIM * 2);
  unsigned short* wvb = (unsigned short*)carve((size_t)(KVH * 128) * DIM * 2);
  unsigned short* wob = (unsigned short*)carve((size_t)DIM * DIM * 2);
  float* Qf = (float*)carve((size_t)M * DIM * 4);          // also reused as attnb
  float* Kf = (float*)carve((size_t)M * (KVH * 128) * 4);
  float* Vf = (float*)carve((size_t)M * (KVH * 128) * 4);
  unsigned short* Qbf = (unsigned short*)carve((size_t)M * DIM * 2);
  unsigned short* Kbf = (unsigned short*)carve((size_t)M * (KVH * 128) * 2);
  unsigned short* Vbf = (unsigned short*)carve((size_t)M * (KVH * 128) * 2);
  unsigned short* attnb = (unsigned short*)Qf;             // Qf dead after RoPE

  auto cgrid = [](int n) { return dim3((unsigned)((n + 255) / 256)); };

  // 1) f32 -> bf16 casts
  cast_bf16_kernel<<<cgrid(M * DIM), 256, 0, stream>>>(x, xb, M * DIM);
  cast_bf16_kernel<<<cgrid(DIM * DIM), 256, 0, stream>>>(wq, wqb, DIM * DIM);
  cast_bf16_kernel<<<cgrid(512 * DIM), 256, 0, stream>>>(wk, wkb, 512 * DIM);
  cast_bf16_kernel<<<cgrid(512 * DIM), 256, 0, stream>>>(wv, wvb, 512 * DIM);
  cast_bf16_kernel<<<cgrid(DIM * DIM), 256, 0, stream>>>(wo, wob, DIM * DIM);

  // 2) QKV projections (WMMA GEMMs, f32 out)
  gemm_bf16_nt<<<dim3(DIM / 128, M / 16), 32, 0, stream>>>(xb, wqb, Qf, M, DIM, DIM);
  gemm_bf16_nt<<<dim3(512 / 128, M / 16), 32, 0, stream>>>(xb, wkb, Kf, M, 512, DIM);
  gemm_bf16_nt<<<dim3(512 / 128, M / 16), 32, 0, stream>>>(xb, wvb, Vf, M, 512, DIM);

  // 3) RoPE + permute to [B, nh, L, 128] bf16
  rope_perm_kernel<<<cgrid(B * H * L * 128), 256, 0, stream>>>(
      Qf, cosb, sinb, Qbf, L, H, B * H * L * 128, 1);
  rope_perm_kernel<<<cgrid(B * KVH * L * 128), 256, 0, stream>>>(
      Kf, cosb, sinb, Kbf, L, KVH, B * KVH * L * 128, 1);
  rope_perm_kernel<<<cgrid(B * KVH * L * 128), 256, 0, stream>>>(
      Vf, cosb, sinb, Vbf, L, KVH, B * KVH * L * 128, 0);

  // 4) sliding-window flash attention -> attnb [B, L, H*128] bf16
  attn_sw_kernel<<<dim3(L / 16, H, B), 32, 0, stream>>>(Qbf, Kbf, Vbf, attnb, L);

  // 5) output projection -> f32 d_out
  gemm_bf16_nt<<<dim3(DIM / 128, M / 16), 32, 0, stream>>>(attnb, wob, out, M, DIM, DIM);
}
